// Decoder_23991687316084
// MI455X (gfx1250) — matrix operand order repaired and checked by reading, not compiled, not run on hardware
//
#include <hip/hip_runtime.h>
#include <math.h>

// ---- dims ----
#define B_   64
#define TDEC 199
#define TMEM 200
#define V_   8000
#define E_   256
#define R_   1024
#define D_   1024
#define KZ   (E_ + D_ + R_)   // 2304 fused LSTM K
#define NZ   (4 * R_)         // 4096
#define KA   (R_ + R_)        // 2048
#define MKEY (B_ * TMEM)      // 12800 rows for keys GEMM

typedef _Float16 v16h __attribute__((ext_vector_type(16)));
typedef _Float16 v8h  __attribute__((ext_vector_type(8)));
typedef float    v8f  __attribute__((ext_vector_type(8)));

// ---------------------------------------------------------------------------
// f16 WMMA GEMM with 64x32 macro-tile per wave (4 M-tiles x 2 N-tiles).
// C[M x N] = A[M x K] * Bt[N x K]^T (+bias). A row-major f16 (lda elems),
// Bt stored [N][K] f16 so B fragments are contiguous 32B loads.
// 8 v_wmma_f32_16x16x32_f16 per K-step fed by 12 b128 loads.
// ---------------------------------------------------------------------------
__global__ __launch_bounds__(256)
void gemm_f16_wmma(const _Float16* __restrict__ A, int lda,
                   const _Float16* __restrict__ Bt, int K,
                   float* __restrict__ C32, long long ldc32,
                   _Float16* __restrict__ C16, int ldc16,
                   const float* __restrict__ bias,
                   int M64, int N32)            // macro grid: (M/64) x (N/32)
{
    const int wave  = threadIdx.x >> 5;
    const int tile  = blockIdx.x * (blockDim.x >> 5) + wave;
    if (tile >= M64 * N32) return;              // wave-uniform exit
    const int mm   = tile / N32;                // 64-row strip
    const int nn   = tile % N32;                // 32-col strip
    const int lane = threadIdx.x & 31;
    const int half = lane >> 4;                 // 0: lanes 0-15, 1: 16-31
    const int lrow = lane & 15;

    // A fragment base: row = mm*64 + mi*16 + lrow;
    // lane-half selects K 0..15 / 16..31
    const _Float16* arow0 = A + (size_t)(mm * 64 + lrow) * lda + half * 16;
    const size_t astride  = (size_t)16 * lda;
    // B fragment base: col = nn*32 + ni*16 + lrow; lane-half selects K 0..15 / 16..31
    const _Float16* brow0 = Bt + (size_t)(nn * 32 + lrow) * K + half * 16;
    const size_t bstride  = (size_t)16 * K;

    v8f acc[4][2] = {};
    for (int k0 = 0; k0 < K; k0 += 32) {
        __builtin_prefetch(brow0 + k0 + 256, 0, 1);
        v16h af[4], bf[2];
#pragma unroll
        for (int mi = 0; mi < 4; ++mi) {
            const _Float16* ap = arow0 + (size_t)mi * astride + k0;
            v8h a0 = *(const v8h*)(ap);          // K = k0 + half*16 + 0..7
            v8h a1 = *(const v8h*)(ap + 8);      // K = k0 + half*16 + 8..15
            af[mi] = __builtin_shufflevector(a0, a1, 0,1,2,3,4,5,6,7,
                                                     8,9,10,11,12,13,14,15);
        }
#pragma unroll
        for (int ni = 0; ni < 2; ++ni)
            bf[ni] = *(const v16h*)(brow0 + (size_t)ni * bstride + k0);
#pragma unroll
        for (int mi = 0; mi < 4; ++mi)
#pragma unroll
            for (int ni = 0; ni < 2; ++ni)
                acc[mi][ni] = __builtin_amdgcn_wmma_f32_16x16x32_f16(
                    false, af[mi], false, bf[ni], (short)0, acc[mi][ni],
                    false, false);
    }

#pragma unroll
    for (int ni = 0; ni < 2; ++ni) {
        const int col = nn * 32 + ni * 16 + lrow;
        const float bv = bias ? bias[col] : 0.0f;
#pragma unroll
        for (int mi = 0; mi < 4; ++mi) {
#pragma unroll
            for (int i = 0; i < 8; ++i) {
                const int row = mm * 64 + mi * 16 + half * 8 + i;  // ISA C/D layout
                const float v = acc[mi][ni][i] + bv;
                if (C32) C32[(size_t)row * ldc32 + col] = v;
                if (C16) C16[(size_t)row * ldc16 + col] = (_Float16)v;
            }
        }
    }
}

// ---- one-time weight conversion ----
// dst[n*K+k] = (k<1280 ? W_lstm[k][n] : U_lstm[k-1280][n]), f32->f16
__global__ void build_wz(const float* __restrict__ Wl, const float* __restrict__ Ul,
                         _Float16* __restrict__ dst)
{
    int idx = blockIdx.x * blockDim.x + threadIdx.x;
    if (idx >= NZ * KZ) return;
    int n = idx / KZ, k = idx % KZ;
    float v = (k < (E_ + D_)) ? Wl[(size_t)k * NZ + n]
                              : Ul[(size_t)(k - (E_ + D_)) * NZ + n];
    dst[idx] = (_Float16)v;
}

// src [K][N] f32 -> dst [N][K] f16
__global__ void transpose_to_f16(const float* __restrict__ src,
                                 _Float16* __restrict__ dst, int K, int N)
{
    int idx = blockIdx.x * blockDim.x + threadIdx.x;
    if (idx >= K * N) return;
    int n = idx / K, k = idx % K;
    dst[idx] = (_Float16)src[(size_t)k * N + n];
}

__global__ void to_f16(const float* __restrict__ s, _Float16* __restrict__ d, int n)
{
    int i = blockIdx.x * blockDim.x + threadIdx.x;
    if (i < n) d[i] = (_Float16)s[i];
}

// ---- per-step kernels ----
// x_t embedding lookup -> cols [0,256) of fused activation row a16[b][2304]
__global__ void embed_step(const int* __restrict__ tok, const float* __restrict__ emb,
                           _Float16* __restrict__ a16, int t)
{
    int idx = blockIdx.x * blockDim.x + threadIdx.x;   // 64*256
    int b = idx >> 8, e = idx & 255;
    int w = tok[b * TDEC + t];
    a16[(size_t)b * KZ + e] = (_Float16)emb[(size_t)w * E_ + e];
}

__device__ __forceinline__ float sigm(float x) { return 1.0f / (1.0f + expf(-x)); }

// gates: z[64][4096] (bias already fused) -> update c,h; mirror h into f16 slots
__global__ void lstm_gates(const float* __restrict__ z,
                           float* __restrict__ h, float* __restrict__ c,
                           _Float16* __restrict__ a16, _Float16* __restrict__ ha16)
{
    int idx = blockIdx.x * blockDim.x + threadIdx.x;   // 64*1024
    if (idx >= B_ * R_) return;
    int b = idx / R_, n = idx % R_;
    const float* zb = z + (size_t)b * NZ;
    float zi = zb[n], zf = zb[n + R_], zg = zb[n + 2 * R_], zo = zb[n + 3 * R_];
    float cn = sigm(zf) * c[idx] + sigm(zi) * tanhf(zg);
    float hn = sigm(zo) * tanhf(cn);
    c[idx] = cn;
    h[idx] = hn;
    a16[(size_t)b * KZ + (E_ + D_) + n] = (_Float16)hn;  // h slot of next A row
    ha16[(size_t)b * KA + n]            = (_Float16)hn;  // h slot of [h;ctx]
}

// Luong attention: one block per batch row. score -> softmax -> context(f16).
// h[b] is staged into LDS via CDNA5 async global->LDS copies (ASYNCcnt).
__global__ __launch_bounds__(256)
void attention(const float* __restrict__ h, const float* __restrict__ keys,
               const float* __restrict__ memory, _Float16* __restrict__ ha16)
{
    __shared__ float hs[R_];
    __shared__ float sc[256];
    __shared__ float red[256];
    const int b = blockIdx.x, tid = threadIdx.x;

    // async global -> LDS stage of h[b] (4 x b32 per lane), then wait + barrier
    for (int i = tid; i < R_; i += 256) {
        unsigned lds_off = (unsigned)(unsigned long long)(&hs[i]);
        unsigned long long ga = (unsigned long long)(h + (size_t)b * R_ + i);
        asm volatile("global_load_async_to_lds_b32 %0, %1, off"
                     :: "v"(lds_off), "v"(ga) : "memory");
    }
    asm volatile("s_wait_asynccnt 0x0" ::: "memory");
    __syncthreads();

    float s = -3.402823466e38f;
    if (tid < TMEM) {
        const float* kr = keys + (size_t)(b * TMEM + tid) * D_;
        float acc = 0.f;
        for (int d = 0; d < D_; ++d) acc += hs[d] * kr[d];
        s = acc;
    }
    red[tid] = s;
    __syncthreads();
    for (int off = 128; off > 0; off >>= 1) {
        if (tid < off) red[tid] = fmaxf(red[tid], red[tid + off]);
        __syncthreads();
    }
    const float mx = red[0];
    __syncthreads();
    float e = (tid < TMEM) ? expf(s - mx) : 0.f;
    red[tid] = e;
    __syncthreads();
    for (int off = 128; off > 0; off >>= 1) {
        if (tid < off) red[tid] += red[tid + off];
        __syncthreads();
    }
    const float Z = red[0];
    __syncthreads();
    sc[tid] = e / Z;
    __syncthreads();

    for (int r = tid; r < R_; r += 256) {
        const float* mb = memory + (size_t)b * TMEM * R_ + r;
        float acc = 0.f;
        for (int t = 0; t < TMEM; ++t) acc += sc[t] * mb[(size_t)t * R_];
        ha16[(size_t)b * KA + R_ + r] = (_Float16)acc;   // context slot
    }
}

// ---------------------------------------------------------------------------
extern "C" void kernel_launch(void* const* d_in, const int* in_sizes, int n_in,
                              void* d_out, int out_size, void* d_ws, size_t ws_size,
                              hipStream_t stream)
{
    const int*   input_batch = (const int*)  d_in[0];
    const float* memory      = (const float*)d_in[1];
    const float* emb         = (const float*)d_in[2];
    const float* Wm          = (const float*)d_in[3];
    const float* Wa          = (const float*)d_in[4];
    const float* Wl          = (const float*)d_in[5];
    const float* Ul          = (const float*)d_in[6];
    const float* bl          = (const float*)d_in[7];
    const float* Wd          = (const float*)d_in[8];
    const float* bd          = (const float*)d_in[9];
    float* out = (float*)d_out;

    // ---- workspace carve-up (256B aligned) ----
    char* cur = (char*)d_ws;
    auto alloc = [&](size_t bytes) {
        char* p = cur; cur += (bytes + 255) & ~(size_t)255; return p;
    };
    _Float16* Wz16  = (_Float16*)alloc((size_t)NZ * KZ * 2);      // 18.9 MB
    _Float16* Wa16  = (_Float16*)alloc((size_t)D_ * KA * 2);      //  4.2 MB
    _Float16* Wd16  = (_Float16*)alloc((size_t)V_ * R_ * 2);      // 16.4 MB
    _Float16* Wm16  = (_Float16*)alloc((size_t)D_ * R_ * 2);      //  2.1 MB
    _Float16* mem16 = (_Float16*)alloc((size_t)MKEY * R_ * 2);    // 26.2 MB
    float*    keys  = (float*)   alloc((size_t)MKEY * D_ * 4);    // 52.4 MB
    float*    z     = (float*)   alloc((size_t)B_ * NZ * 4);
    float*    h     = (float*)   alloc((size_t)B_ * R_ * 4);
    float*    c     = (float*)   alloc((size_t)B_ * R_ * 4);
    _Float16* a16   = (_Float16*)alloc((size_t)B_ * KZ * 2);      // [x;attn;h]
    _Float16* ha16  = (_Float16*)alloc((size_t)B_ * KA * 2);      // [h;context]

    auto cdiv = [](long long a, long long b) { return (int)((a + b - 1) / b); };

    // ---- one-time: f16 weight prep + keys GEMM ----
    build_wz<<<cdiv((long long)NZ * KZ, 256), 256, 0, stream>>>(Wl, Ul, Wz16);
    transpose_to_f16<<<cdiv((long long)KA * D_, 256), 256, 0, stream>>>(Wa, Wa16, KA, D_);
    transpose_to_f16<<<cdiv((long long)R_ * V_, 256), 256, 0, stream>>>(Wd, Wd16, R_, V_);
    transpose_to_f16<<<cdiv((long long)R_ * D_, 256), 256, 0, stream>>>(Wm, Wm16, R_, D_);
    to_f16<<<cdiv((long long)MKEY * R_, 256), 256, 0, stream>>>(memory, mem16, MKEY * R_);

    // keys = memory @ Wm  (M=12800, K=1024, N=1024) -> macro grid 200 x 32
    gemm_f16_wmma<<<cdiv((MKEY / 64) * (D_ / 32), 8), 256, 0, stream>>>(
        mem16, R_, Wm16, R_, keys, D_, nullptr, 0, nullptr, MKEY / 64, D_ / 32);

    // zero initial state (h0=c0=0, attn0=0 -> zero whole a16 / ha16)
    hipMemsetAsync(h,    0, (size_t)B_ * R_ * 4, stream);
    hipMemsetAsync(c,    0, (size_t)B_ * R_ * 4, stream);
    hipMemsetAsync(a16,  0, (size_t)B_ * KZ * 2, stream);
    hipMemsetAsync(ha16, 0, (size_t)B_ * KA * 2, stream);

    // ---- 199 sequential decode steps ----
    for (int t = 0; t < TDEC; ++t) {
        embed_step<<<B_, 256, 0, stream>>>(input_batch, emb, a16, t);

        // z = [x;attn;h] @ [W_lstm;U_lstm] + b   (M=64, K=2304, N=4096)
        gemm_f16_wmma<<<cdiv(NZ / 32, 8), 256, 0, stream>>>(
            a16, KZ, Wz16, KZ, z, NZ, nullptr, 0, bl, 1, NZ / 32);

        lstm_gates<<<cdiv(B_ * R_, 256), 256, 0, stream>>>(z, h, c, a16, ha16);

        attention<<<B_, 256, 0, stream>>>(h, keys, memory, ha16);

        // attn = [h;ctx] @ Wa  -> f16 directly into attn slot of a16 (cols 256..1279)
        gemm_f16_wmma<<<cdiv(D_ / 32, 8), 256, 0, stream>>>(
            ha16, KA, Wa16, KA, nullptr, 0, a16 + E_, KZ, nullptr, 1, D_ / 32);

        // logits = attn @ Wd + bd -> out[:, t, :]   (M=64, K=1024, N=8000)
        gemm_f16_wmma<<<cdiv(V_ / 32, 8), 256, 0, stream>>>(
            a16 + E_, KZ, Wd16, R_, out + (size_t)t * V_, (long long)TDEC * V_,
            nullptr, 0, bd, 1, V_ / 32);
    }

    // ---- tail outputs: final h then c ----
    float* tail = out + (size_t)B_ * TDEC * V_;
    hipMemcpyAsync(tail,            h, (size_t)B_ * R_ * 4, hipMemcpyDeviceToDevice, stream);
    hipMemcpyAsync(tail + B_ * R_,  c, (size_t)B_ * R_ * 4, hipMemcpyDeviceToDevice, stream);
}
